// GATv2Encoder_38156489458108
// MI455X (gfx1250) — compile-verified, matrix operand order checked by reading, not run on hardware
//
#include <hip/hip_runtime.h>
#include <hip/hip_bf16.h>
#include <math.h>

#define N_NODES 50000
#define N_EDGES 800000
#define DCH 128
#define NH 8
#define CPH 16
#define NLAYERS 2
#define NEG_SLOPE 0.2f
#define LN_EPS 1e-12f

typedef __attribute__((ext_vector_type(16))) __bf16        v16bf;
typedef __attribute__((ext_vector_type(8)))  float         v8f;
typedef __attribute__((ext_vector_type(4)))  unsigned int  v4u;

union V16BF { v16bf v; __bf16 e[16]; v4u q[2]; };

#define WPACK_ELEMS (8 * 4 * 32 * 16)   // colTile x kBlock x lane x 16 bf16

__device__ __forceinline__ int f2ord(float f) {
  int i = __float_as_int(f);
  return (i >= 0) ? i : (i ^ 0x7fffffff);
}
__device__ __forceinline__ float ord2f(int i) {
  return __int_as_float((i >= 0) ? i : (i ^ 0x7fffffff));
}

// ---------------------------------------------------------------------------
// Pack a 128x128 fp32 weight matrix into the WMMA bf16 B-operand lane layout,
// split precision (hi + residual lo).  Done once per matrix per launch; the
// GEMM then loads B with two contiguous 16B loads per operand, no conversions.
// Layout: idx = ((colTile*4 + kb)*32 + lane)*16 + e, where for lane half h,
// element e maps to k = kb*32 + h*8 + e (+8 more for e>=8), n = colTile*16+l15.
// ---------------------------------------------------------------------------
__global__ void __launch_bounds__(256)
pack_w_bf16(const float* __restrict__ Wm, __bf16* __restrict__ Bhi,
            __bf16* __restrict__ Blo)
{
  int idx = blockIdx.x * 256 + threadIdx.x;
  if (idx >= WPACK_ELEMS) return;
  int e    = idx & 15;
  int lane = (idx >> 4) & 31;
  int kb   = (idx >> 9) & 3;
  int ct   = idx >> 11;
  int half = lane >> 4, l15 = lane & 15;
  int k = kb * 32 + half * 8 + e + ((e >= 8) ? 8 : 0);
  float v = Wm[k * DCH + ct * CPH + l15];
  __bf16 hv = (__bf16)v;
  Bhi[idx] = hv;
  Blo[idx] = (__bf16)(v - (float)hv);
}

// ---------------------------------------------------------------------------
// GEMM: Y[M,128] = X[M,128] @ W (+bias), W pre-packed as bf16 hi/lo operands.
// Split-precision: acc += Ah*Bh + Ah*Bl + Al*Bh (near-fp32 accuracy).
// Block = 256 threads = 8 waves; wave w owns the 16x16 tile at column 16w.
// X tile converted to bf16 hi/lo once per block in LDS (row pad +8 bf16 =>
// 272B row stride => conflict-free 16-lane column reads).
// ---------------------------------------------------------------------------
__global__ void __launch_bounds__(256)
gemm128_wmma_bf16x2(const float* __restrict__ X,
                    const __bf16* __restrict__ Bhi,
                    const __bf16* __restrict__ Blo,
                    const float* __restrict__ bias, float* __restrict__ Y,
                    int Mfixed, const int* __restrict__ Mptr)
{
  int M = Mptr ? *Mptr : Mfixed;
  int row0 = blockIdx.x * 16;
  if (row0 >= M) return;

  int wave = threadIdx.x >> 5;
  int lane = threadIdx.x & 31;
  int half = lane >> 4;        // 0: lanes 0-15, 1: lanes 16-31
  int l15  = lane & 15;

  __shared__ __align__(16) __bf16 xh[16][DCH + 8];
  __shared__ __align__(16) __bf16 xl[16][DCH + 8];
  for (int i = threadIdx.x; i < 16 * DCH; i += 256) {
    int r = i >> 7, c = i & (DCH - 1);
    int gr = row0 + r;
    float v = (gr < M) ? X[(long long)gr * DCH + c] : 0.f;
    __bf16 hv = (__bf16)v;
    xh[r][c] = hv;
    xl[r][c] = (__bf16)(v - (float)hv);
  }
  __syncthreads();

  v8f acc = {};
#pragma unroll
  for (int kb = 0; kb < DCH; kb += 32) {
    V16BF a_hi, a_lo, b_hi, b_lo;
    int o0 = kb + half * 8;          // contiguous runs of the A lane layout
    int o1 = kb + 16 + half * 8;
    a_hi.q[0] = *(const v4u*)&xh[l15][o0];
    a_hi.q[1] = *(const v4u*)&xh[l15][o1];
    a_lo.q[0] = *(const v4u*)&xl[l15][o0];
    a_lo.q[1] = *(const v4u*)&xl[l15][o1];
    const __bf16* bph = Bhi + (size_t)((((wave << 2) + (kb >> 5)) << 5) + lane) * 16;
    const __bf16* bpl = Blo + (size_t)((((wave << 2) + (kb >> 5)) << 5) + lane) * 16;
    b_hi.q[0] = ((const v4u*)bph)[0];
    b_hi.q[1] = ((const v4u*)bph)[1];
    b_lo.q[0] = ((const v4u*)bpl)[0];
    b_lo.q[1] = ((const v4u*)bpl)[1];
    acc = __builtin_amdgcn_wmma_f32_16x16x32_bf16(false, a_hi.v, false, b_hi.v,
                                                  (short)0, acc, false, false);
    acc = __builtin_amdgcn_wmma_f32_16x16x32_bf16(false, a_hi.v, false, b_lo.v,
                                                  (short)0, acc, false, false);
    acc = __builtin_amdgcn_wmma_f32_16x16x32_bf16(false, a_lo.v, false, b_hi.v,
                                                  (short)0, acc, false, false);
  }

  int col0 = wave * 16;
  float bv = bias ? bias[col0 + l15] : 0.f;
  // C/D layout: lanes 0-15 hold M=0..7 in VGPR 0..7; lanes 16-31 hold M=8..15
#pragma unroll
  for (int r = 0; r < 8; ++r) {
    int gm = row0 + half * 8 + r;
    if (gm < M) Y[(long long)gm * DCH + col0 + l15] = acc[r] + bv;
  }
}

// ---------------------------------------------------------------------------
// Per-layer node-state init: agg = bias (broadcast), segment max/sum reset.
// ---------------------------------------------------------------------------
__global__ void __launch_bounds__(256)
init_nodes(float* __restrict__ agg, int* __restrict__ nmax,
           float* __restrict__ nsum, const float* __restrict__ bias)
{
  int idx = blockIdx.x * 256 + threadIdx.x;
  if (idx >= N_NODES * DCH) return;
  int c = idx & (DCH - 1);
  agg[idx] = bias[c];
  if (c < NH) {
    int node = idx >> 7;
    nmax[node * NH + c] = f2ord(__int_as_float(0xff800000u));  // -inf
    nsum[node * NH + c] = 0.f;
  }
}

// ---------------------------------------------------------------------------
// Pass 1: per-edge GATv2 logits + segment max.  One wave per edge; each lane
// handles 4 channels (float4), quad-of-lanes = one head; reduce within quad.
// h and nmax are L2-resident (25.6MB / 1.6MB << 192MB L2).
// ---------------------------------------------------------------------------
__global__ void __launch_bounds__(256)
edge_logits(const float* __restrict__ h, const long long* __restrict__ src,
            const long long* __restrict__ dst, const float* __restrict__ att,
            float* __restrict__ exl, int* __restrict__ nmax)
{
  int e = (blockIdx.x * 256 + threadIdx.x) >> 5;
  if (e >= N_EDGES) return;
  int l = threadIdx.x & 31;
  long long s = src[e], d = dst[e];

  float4 a = ((const float4*)(h + s * DCH))[l];
  float4 b = ((const float4*)(h + d * DCH))[l];
  int head = l >> 2;
  const float* at = att + head * CPH + (l & 3) * 4;

  float v0 = a.x + b.x; v0 = v0 > 0.f ? v0 : v0 * NEG_SLOPE;
  float v1 = a.y + b.y; v1 = v1 > 0.f ? v1 : v1 * NEG_SLOPE;
  float v2 = a.z + b.z; v2 = v2 > 0.f ? v2 : v2 * NEG_SLOPE;
  float v3 = a.w + b.w; v3 = v3 > 0.f ? v3 : v3 * NEG_SLOPE;
  float sum = v0 * at[0] + v1 * at[1] + v2 * at[2] + v3 * at[3];

  sum += __shfl_xor(sum, 1);
  sum += __shfl_xor(sum, 2);
  if ((l & 3) == 0) {
    exl[(long long)e * NH + head] = sum;
    atomicMax(&nmax[d * NH + head], f2ord(sum));
  }
}

// ---------------------------------------------------------------------------
// Pass 2: ex = exp(logit - max[dst]); segment sum.  Thread per (edge, head).
// ---------------------------------------------------------------------------
__global__ void __launch_bounds__(256)
edge_expsum(const long long* __restrict__ dst, float* __restrict__ exl,
            const int* __restrict__ nmax, float* __restrict__ nsum)
{
  int idx = blockIdx.x * 256 + threadIdx.x;
  if (idx >= N_EDGES * NH) return;
  int e = idx >> 3, hh = idx & 7;
  long long d = dst[e];
  float m = ord2f(nmax[d * NH + hh]);
  float ex = __expf(exl[idx] - m);
  exl[idx] = ex;
  atomicAdd(&nsum[d * NH + hh], ex);
}

// ---------------------------------------------------------------------------
// Pass 3: alpha = ex / sum[dst]; agg[dst] += alpha * h[src].  Wave per edge.
// ---------------------------------------------------------------------------
__global__ void __launch_bounds__(256)
edge_aggregate(const float* __restrict__ h, const long long* __restrict__ src,
               const long long* __restrict__ dst, const float* __restrict__ exl,
               const float* __restrict__ nsum, float* __restrict__ agg)
{
  int e = (blockIdx.x * 256 + threadIdx.x) >> 5;
  if (e >= N_EDGES) return;
  int l = threadIdx.x & 31;
  long long s = src[e], d = dst[e];
  int head = l >> 2;
  float alpha = exl[(long long)e * NH + head] / (nsum[d * NH + head] + 1e-16f);
  float4 a = ((const float4*)(h + s * DCH))[l];
  float* o = agg + d * DCH + l * 4;
  atomicAdd(o + 0, a.x * alpha);
  atomicAdd(o + 1, a.y * alpha);
  atomicAdd(o + 2, a.z * alpha);
  atomicAdd(o + 3, a.w * alpha);
}

__global__ void __launch_bounds__(256)
gelu_inplace(float* __restrict__ x, int n)
{
  int i = blockIdx.x * 256 + threadIdx.x;
  if (i >= n) return;
  float v = x[i];
  x[i] = 0.5f * v * (1.f + erff(v * 0.70710678118654752f));
}

// ---------------------------------------------------------------------------
// LayerNorm over 128 channels, wave per row, in-place on d_out.
// ---------------------------------------------------------------------------
__global__ void __launch_bounds__(256)
layernorm_rows(float* __restrict__ y, const float* __restrict__ g,
               const float* __restrict__ b, const int* __restrict__ Mptr)
{
  int M = *Mptr;
  int row = (blockIdx.x * 256 + threadIdx.x) >> 5;
  if (row >= M) return;
  int l = threadIdx.x & 31;
  float4 v = ((const float4*)(y + (long long)row * DCH))[l];
  float s  = v.x + v.y + v.z + v.w;
  float sq = v.x * v.x + v.y * v.y + v.z * v.z + v.w * v.w;
#pragma unroll
  for (int m = 1; m < 32; m <<= 1) {
    s  += __shfl_xor(s,  m);
    sq += __shfl_xor(sq, m);
  }
  float mean = s * (1.f / 128.f);
  float var  = sq * (1.f / 128.f) - mean * mean;
  float rstd = rsqrtf(var + LN_EPS);
  int c = l * 4;
  float4 o;
  o.x = (v.x - mean) * rstd * g[c + 0] + b[c + 0];
  o.y = (v.y - mean) * rstd * g[c + 1] + b[c + 1];
  o.z = (v.z - mean) * rstd * g[c + 2] + b[c + 2];
  o.w = (v.w - mean) * rstd * g[c + 3] + b[c + 3];
  ((float4*)(y + (long long)row * DCH))[l] = o;
}

extern "C" void kernel_launch(void* const* d_in, const int* in_sizes, int n_in,
                              void* d_out, int out_size, void* d_ws, size_t ws_size,
                              hipStream_t stream)
{
  const float*     x    = (const float*)d_in[0];
  const long long* ei   = (const long long*)d_in[1];   // int64 per reference
  const float*     W    = (const float*)d_in[2];
  const float*     att  = (const float*)d_in[3];
  const float*     bias = (const float*)d_in[4];
  const float*     outW = (const float*)d_in[5];
  const float*     outb = (const float*)d_in[6];
  const float*     lng  = (const float*)d_in[7];
  const float*     lnb  = (const float*)d_in[8];
  const int*       ntrg = (const int*)d_in[9];          // low word of int64 (LE)

  const long long* src  = ei;
  const long long* dstp = ei + N_EDGES;

  float* h    = (float*)d_ws;                            // [N,128]
  float* agg  = h   + (size_t)N_NODES * DCH;             // [N,128]
  float* exl  = agg + (size_t)N_NODES * DCH;             // [E,8]
  int*   nmax = (int*)(exl + (size_t)N_EDGES * NH);      // [N,8]
  float* nsum = (float*)(nmax + (size_t)N_NODES * NH);   // [N,8]
  __bf16* wp  = (__bf16*)(nsum + (size_t)N_NODES * NH);  // packed weights
  __bf16* Whi[NLAYERS + 1];
  __bf16* Wlo[NLAYERS + 1];
  for (int i = 0; i < NLAYERS + 1; ++i) {
    Whi[i] = wp + (size_t)(2 * i)     * WPACK_ELEMS;
    Wlo[i] = wp + (size_t)(2 * i + 1) * WPACK_ELEMS;
  }

  dim3 blk(256);
  int gemmGrid = N_NODES / 16;                  // 3125 (50000 % 16 == 0)
  int nodeGrid = (N_NODES * DCH + 255) / 256;
  int edgeGrid = (N_EDGES + 7) / 8;             // wave per edge, 8 waves/block
  int ehGrid   = (N_EDGES * NH + 255) / 256;
  int packGrid = WPACK_ELEMS / 256;             // 64

  // Pack all weight matrices once (layers 0,1 + output head).
  for (int i = 0; i < NLAYERS; ++i)
    pack_w_bf16<<<packGrid, blk, 0, stream>>>(W + (size_t)i * DCH * DCH,
                                              Whi[i], Wlo[i]);
  pack_w_bf16<<<packGrid, blk, 0, stream>>>(outW, Whi[NLAYERS], Wlo[NLAYERS]);

  const float* xin = x;
  for (int i = 0; i < NLAYERS; ++i) {
    gemm128_wmma_bf16x2<<<gemmGrid, blk, 0, stream>>>(
        xin, Whi[i], Wlo[i], nullptr, h, N_NODES, nullptr);
    init_nodes<<<nodeGrid, blk, 0, stream>>>(agg, nmax, nsum, bias + i * DCH);
    edge_logits<<<edgeGrid, blk, 0, stream>>>(h, src, dstp,
                                              att + i * NH * CPH, exl, nmax);
    edge_expsum<<<ehGrid, blk, 0, stream>>>(dstp, exl, nmax, nsum);
    edge_aggregate<<<edgeGrid, blk, 0, stream>>>(h, src, dstp, exl, nsum, agg);
    if (i != NLAYERS - 1)
      gelu_inplace<<<nodeGrid, blk, 0, stream>>>(agg, N_NODES * DCH);
    xin = agg;
  }

  // Output head: Linear (+bias) via WMMA into d_out, then LayerNorm in-place.
  gemm128_wmma_bf16x2<<<gemmGrid, blk, 0, stream>>>(
      agg, Whi[NLAYERS], Wlo[NLAYERS], outb, (float*)d_out, N_NODES, ntrg);
  layernorm_rows<<<(N_NODES + 7) / 8, blk, 0, stream>>>(
      (float*)d_out, lng, lnb, ntrg);
}
